// ExpertNetwork_56745107915273
// MI455X (gfx1250) — matrix operand order repaired and checked by reading, not compiled, not run on hardware
//
#include <hip/hip_runtime.h>
#include <hip/hip_bf16.h>
#include <math.h>

// ---------------------------------------------------------------------------
// Types for CDNA5 WMMA (wave32, v_wmma_f32_16x16x32_bf16) + TDM descriptors
// ---------------------------------------------------------------------------
typedef __bf16 bf16_t;
typedef __attribute__((ext_vector_type(16))) __bf16 v16bf;
typedef __attribute__((ext_vector_type(8)))  float  v8f;
typedef __attribute__((ext_vector_type(4)))  unsigned int u32x4;
typedef __attribute__((ext_vector_type(4)))  int i32x4;
typedef __attribute__((ext_vector_type(8)))  int i32x8;

#define DEV __device__ __forceinline__

#if __has_builtin(__builtin_amdgcn_tensor_load_to_lds) && \
    __has_builtin(__builtin_amdgcn_s_wait_tensorcnt)
#define HAVE_TDM 1
#else
#define HAVE_TDM 0
#endif

constexpr int Lc = 2, Dc = 1024, Hc = 16, Fc = 4096, Bc = 16, Sc = 512, DHc = 64;
constexpr int Mrows = Bc * Sc; // 8192

union Frag { v16bf v; uint4 q[2]; };

// A-matrix fragment (16x32 bf16, MxK). Per ISA table:
// lanes 0-15 (m): V0..3 hold K=0..7, V4..7 hold K=16..23
// lanes 16-31  : V0..3 hold K=8..15, V4..7 hold K=24..31
DEV void loadFragA(Frag& f, const bf16_t* rowBase, int ldElems, int kElem, int lane) {
  const int m16 = lane & 15, hi = lane >> 4;
  const bf16_t* p = rowBase + (size_t)m16 * ldElems + kElem + hi * 8;
  f.q[0] = *(const uint4*)(p);
  f.q[1] = *(const uint4*)(p + 16);
}

// B-matrix fragment (32x16 bf16, KxN), source stored as [N,K] row-major.
// lanes 0-15 (n): V0..3 hold K=0..7, V4..7 hold K=16..23; lanes 16-31: K=8..15, K=24..31.
DEV void loadFragB(Frag& f, const bf16_t* rowBase, int ldElems, int kElem, int lane) {
  const int n16 = lane & 15, hi = lane >> 4;
  const bf16_t* p = rowBase + (size_t)n16 * ldElems + kElem + hi * 8;
  f.q[0] = *(const uint4*)(p);
  f.q[1] = *(const uint4*)(p + 16);
}

DEV v8f wmma_bf16(const Frag& a, const Frag& b, v8f c) {
  return __builtin_amdgcn_wmma_f32_16x16x32_bf16(false, a.v, false, b.v,
                                                 (short)0, c, false, false);
}

#if HAVE_TDM
// Tensor Data Mover: DMA a [128 rows x 32 bf16] tile (row stride K elems) from
// global into LDS with hardware row padding: every 16 DWORDs (64B row) insert
// 4 DWORDs (16B) -> LDS row stride 80B, matching the 40-elem fragment layout.
DEV void tdm_load_tile_128x32(const bf16_t* gbase, int rowBase, int K, int k0,
                              unsigned int ldsAddr) {
  const unsigned long long ga =
      (unsigned long long)(uintptr_t)gbase + ((unsigned long long)rowBase * K + k0) * 2u;
  u32x4 g0;
  g0[0] = 1u;                                   // count=1, load, user mode
  g0[1] = ldsAddr;                              // lds_addr (bytes)
  g0[2] = (unsigned int)ga;                     // global_addr[31:0]
  g0[3] = ((unsigned int)(ga >> 32) & 0x01FFFFFFu) | (2u << 30); // [56:32] | type=2
  i32x8 g1;
  g1[0] = (1 << 16)    // data_size = 2 bytes
        | (1 << 20)    // pad_enable
        | (3 << 22)    // pad_interval: 16 DWORDs (64B) between pads
        | (3 << 25);   // pad_amount: 4 DWORDs (16B)
  g1[1] = (int)(((unsigned)K & 0xFFFFu) << 16);                 // tensor_dim0 lo16
  g1[2] = (int)(((unsigned)K >> 16) | (8192u << 16));           // dim0 hi16 | tensor_dim1 lo16
  g1[3] = (int)(32u << 16);                                     // tensor_dim1 hi16 | tile_dim0=32
  g1[4] = (int)128u;                                            // tile_dim1=128, tile_dim2=0
  g1[5] = (int)K;                                               // tensor_dim0_stride lo32
  g1[6] = 0;                                                    // stride0 hi16 | stride1 lo16
  g1[7] = 0;
  i32x4 gz = {0, 0, 0, 0};
#if __clang_major__ >= 23
  i32x8 gz8 = {0, 0, 0, 0, 0, 0, 0, 0};
  __builtin_amdgcn_tensor_load_to_lds(g0, g1, gz, gz, gz8, 0);
#else
  __builtin_amdgcn_tensor_load_to_lds(g0, g1, gz, gz, 0);
#endif
}
#endif

// ---------------------------------------------------------------------------
// GEMM: C[M,N] = A[M,K] (bf16) @ W[N,K]^T (bf16) + bias, epilogue variants.
// Block tile 128x128x32, 8 waves, each wave 32x64 (2x4 WMMA accs).
// Double-buffered TDM pipeline: wave0 DMAs tile k+1 while all compute tile k.
// EPI: 0 = fp32 out, 1 = bf16 out, 2 = GELU(exact) -> bf16 out
// ---------------------------------------------------------------------------
template <int EPI>
__global__ __launch_bounds__(256) void gemm_kernel(const bf16_t* __restrict__ A,
                                                   const bf16_t* __restrict__ W,
                                                   const float* __restrict__ bias,
                                                   void* __restrict__ outp,
                                                   int M, int N, int K) {
  __shared__ bf16_t As[2][128 * 40]; // 32 bf16 + 8 pad per row (80B, 16B aligned)
  __shared__ bf16_t Bs[2][128 * 40];

  const int t = threadIdx.x;
  const int lane = t & 31, wv = t >> 5;
  const int wm = wv & 3, wn = wv >> 2; // 4x2 wave grid
  const int mBase = blockIdx.y * 128, nBase = blockIdx.x * 128;

  v8f acc[2][4];
#pragma unroll
  for (int i = 0; i < 2; i++)
#pragma unroll
    for (int j = 0; j < 4; j++) acc[i][j] = v8f{};

  const int nk = K >> 5;

#if HAVE_TDM
  if (wv == 0) {
    tdm_load_tile_128x32(A, mBase, K, 0, (unsigned int)(uintptr_t)&As[0][0]);
    tdm_load_tile_128x32(W, nBase, K, 0, (unsigned int)(uintptr_t)&Bs[0][0]);
  }
  for (int kt = 0; kt < nk; kt++) {
    const int cur = kt & 1;
    __syncthreads(); // everyone done reading buf[cur^1] -> safe to overwrite
    if (wv == 0) {
      if (kt + 1 < nk) {
        tdm_load_tile_128x32(A, mBase, K, (kt + 1) * 32,
                             (unsigned int)(uintptr_t)&As[cur ^ 1][0]);
        tdm_load_tile_128x32(W, nBase, K, (kt + 1) * 32,
                             (unsigned int)(uintptr_t)&Bs[cur ^ 1][0]);
        __builtin_amdgcn_s_wait_tensorcnt((short)2); // tile kt landed (in-order TDM)
      } else {
        __builtin_amdgcn_s_wait_tensorcnt((short)0);
      }
    }
    __syncthreads(); // tile kt visible to all waves

    Frag a0, a1;
    loadFragA(a0, As[cur] + (wm * 32) * 40, 40, 0, lane);
    loadFragA(a1, As[cur] + (wm * 32 + 16) * 40, 40, 0, lane);
#pragma unroll
    for (int j = 0; j < 4; j++) {
      Frag b;
      loadFragB(b, Bs[cur] + (wn * 64 + j * 16) * 40, 40, 0, lane);
      acc[0][j] = wmma_bf16(a0, b, acc[0][j]);
      acc[1][j] = wmma_bf16(a1, b, acc[1][j]);
    }
  }
#else
  for (int kt = 0; kt < nk; kt++) {
    const int k0 = kt * 32;
#pragma unroll
    for (int i = 0; i < 2; i++) {
      const int c = t + i * 256;
      const int row = c >> 2, cc = c & 3;
      const bf16_t* ga = A + (size_t)(mBase + row) * K + k0 + cc * 8;
      const bf16_t* gw = W + (size_t)(nBase + row) * K + k0 + cc * 8;
      *(uint4*)(As[0] + row * 40 + cc * 8) = *(const uint4*)ga;
      *(uint4*)(Bs[0] + row * 40 + cc * 8) = *(const uint4*)gw;
      if (k0 + 32 < K) {
        __builtin_prefetch(ga + 32, 0, 1);
        __builtin_prefetch(gw + 32, 0, 1);
      }
    }
    __syncthreads();
    Frag a0, a1;
    loadFragA(a0, As[0] + (wm * 32) * 40, 40, 0, lane);
    loadFragA(a1, As[0] + (wm * 32 + 16) * 40, 40, 0, lane);
#pragma unroll
    for (int j = 0; j < 4; j++) {
      Frag b;
      loadFragB(b, Bs[0] + (wn * 64 + j * 16) * 40, 40, 0, lane);
      acc[0][j] = wmma_bf16(a0, b, acc[0][j]);
      acc[1][j] = wmma_bf16(a1, b, acc[1][j]);
    }
    __syncthreads();
  }
#endif

  // epilogue: C layout -> lane n=lane&15; VGPR r -> m = r + 8*(lane>>4)
  const int m16 = lane & 15, hi = lane >> 4;
#pragma unroll
  for (int i = 0; i < 2; i++) {
#pragma unroll
    for (int j = 0; j < 4; j++) {
      const int n = nBase + wn * 64 + j * 16 + m16;
      const float bv = bias[n];
#pragma unroll
      for (int r = 0; r < 8; r++) {
        const int m = mBase + wm * 32 + i * 16 + r + 8 * hi;
        float val = acc[i][j][r] + bv;
        if (EPI == 2) val = 0.5f * val * (1.0f + erff(val * 0.70710678118654752f));
        if (EPI == 0)
          ((float*)outp)[(size_t)m * N + n] = val;
        else
          ((bf16_t*)outp)[(size_t)m * N + n] = (bf16_t)val;
      }
    }
  }
}

// ---------------------------------------------------------------------------
// De-stationary attention: one block = (b, h, 32-query tile).
// scores = (1/sqrt(DH)) * (tau_b * QK^T + delta[key]); softmax; P @ V.
// ---------------------------------------------------------------------------
constexpr int SC_LD = 516;  // fp32 score row stride (pad)
constexpr int PB_LD = 520;  // bf16 prob row stride  (pad, 1040B = 16B-mult)
constexpr size_t ATT_SMEM =
    (size_t)32 * SC_LD * 4 + (size_t)32 * PB_LD * 2 + 256 * 4 * 2 + 32 * 4;

__global__ __launch_bounds__(256) void attn_kernel(const bf16_t* __restrict__ Q,
                                                   const bf16_t* __restrict__ Kt,
                                                   const bf16_t* __restrict__ Vt,
                                                   const float* __restrict__ tau,
                                                   const float* __restrict__ delta,
                                                   bf16_t* __restrict__ outp) {
  extern __shared__ char smem[];
  float* sc      = (float*)smem;                                   // 32 x 516 f32
  bf16_t* pb     = (bf16_t*)(smem + (size_t)32 * SC_LD * 4);       // 32 x 520 bf16
  float* redA    = (float*)(smem + (size_t)32 * SC_LD * 4 + (size_t)32 * PB_LD * 2);
  float* redB    = redA + 256;
  float* rowinv  = redB + 256;

  const int t = threadIdx.x, lane = t & 31, wv = t >> 5;
  const int bid = blockIdx.x;
  const int qt = bid & 15, h = (bid >> 4) & 15, b = bid >> 8;
  const int m16 = lane & 15, hi = lane >> 4;

  const float sca = tau[b] * 0.125f; // tau_b / sqrt(64)
  const float scd = 0.125f;          // 1 / sqrt(64)

  // ---- phase 1: scaled scores via WMMA (each wave: 16 rows x 128 keys) ----
  {
    const int mstrip = (wv & 1) * 16;
    const int colq = (wv >> 1) * 128;
    const bf16_t* qrow = Q + ((size_t)(b * Sc + qt * 32 + mstrip)) * Dc + h * DHc;
    Frag a0, a1;
    loadFragA(a0, qrow, Dc, 0, lane);
    loadFragA(a1, qrow, Dc, 32, lane);
#pragma unroll
    for (int ct = 0; ct < 8; ct++) {
      const int key0 = colq + ct * 16;
      const bf16_t* krow = Kt + ((size_t)(b * Sc + key0)) * Dc + h * DHc;
      Frag b0, b1;
      loadFragB(b0, krow, Dc, 0, lane);
      loadFragB(b1, krow, Dc, 32, lane);
      v8f a = v8f{};
      a = wmma_bf16(a0, b0, a);
      a = wmma_bf16(a1, b1, a);
      const float dl = delta[b * Sc + key0 + m16] * scd;
#pragma unroll
      for (int r = 0; r < 8; r++)
        sc[(mstrip + r + 8 * hi) * SC_LD + key0 + m16] = sca * a[r] + dl;
    }
  }
  __syncthreads();

  // ---- phase 2: softmax (store unnormalized exp as bf16; keep 1/sum) ----
  {
    const int row = t >> 3, seg = t & 7;
    const float* srow = sc + row * SC_LD + seg * 64;
    float lm = -1e30f;
    for (int c = 0; c < 64; c++) lm = fmaxf(lm, srow[c]);
    redA[t] = lm;
    __syncthreads();
    float rm = -1e30f;
#pragma unroll
    for (int i = 0; i < 8; i++) rm = fmaxf(rm, redA[(row << 3) + i]);
    bf16_t* prow = pb + row * PB_LD + seg * 64;
    float ls = 0.f;
    for (int c = 0; c < 64; c++) {
      float e = __expf(srow[c] - rm);
      prow[c] = (bf16_t)e;
      ls += e;
    }
    redB[t] = ls;
    __syncthreads();
    if (seg == 0) {
      float s = 0.f;
#pragma unroll
      for (int i = 0; i < 8; i++) s += redB[(row << 3) + i];
      rowinv[row] = 1.f / s;
    }
  }
  __syncthreads();

  // ---- phase 3: out = (P @ V) * (1/sum) via WMMA (each wave: 16x16 tile) ----
  {
    const int m0 = (wv & 1) * 16;
    const int d0 = (wv >> 1) * 16;
    const bf16_t* vrow = Vt + ((size_t)((b * Hc + h) * DHc + d0)) * Sc;
    v8f a = v8f{};
    for (int st = 0; st < 16; st++) {
      Frag af, bf_;
      loadFragA(af, pb + m0 * PB_LD, PB_LD, st * 32, lane);
      loadFragB(bf_, vrow, Sc, st * 32, lane);
      a = wmma_bf16(af, bf_, a);
    }
#pragma unroll
    for (int r = 0; r < 8; r++) {
      const int row = m0 + r + 8 * hi;
      const float val = a[r] * rowinv[row];
      outp[((size_t)(b * Sc + qt * 32 + row)) * Dc + h * DHc + d0 + m16] = (bf16_t)val;
    }
  }
}

// ---------------------------------------------------------------------------
// LayerNorm over D=1024 with fused residual add; writes fp32 + bf16 copies.
// ---------------------------------------------------------------------------
__global__ __launch_bounds__(256) void ln_kernel(const float* __restrict__ y,
                                                 const float* __restrict__ resid,
                                                 const float* __restrict__ g,
                                                 const float* __restrict__ be,
                                                 float* __restrict__ xo,
                                                 bf16_t* __restrict__ xb) {
  __shared__ float s1[256], s2[256];
  const int t = threadIdx.x;
  const size_t base = (size_t)blockIdx.x * Dc;
  float v[4], a = 0.f, sq = 0.f;
#pragma unroll
  for (int i = 0; i < 4; i++) {
    const int c = i * 256 + t;
    float val = y[base + c];
    if (resid) val += resid[base + c];
    v[i] = val;
    a += val;
    sq += val * val;
  }
  s1[t] = a; s2[t] = sq;
  __syncthreads();
  for (int off = 128; off > 0; off >>= 1) {
    if (t < off) { s1[t] += s1[t + off]; s2[t] += s2[t + off]; }
    __syncthreads();
  }
  const float mean = s1[0] * (1.f / Dc);
  const float var  = s2[0] * (1.f / Dc) - mean * mean;
  const float rs   = rsqrtf(var + 1e-5f);
#pragma unroll
  for (int i = 0; i < 4; i++) {
    const int c = i * 256 + t;
    const float o = (v[i] - mean) * rs * g[c] + be[c];
    xo[base + c] = o;
    xb[base + c] = (bf16_t)o;
  }
}

// ---------------------------------------------------------------------------
// fp32 -> bf16 conversion and V transpose [B,S,H,DH] -> [B,H,DH,S]
// ---------------------------------------------------------------------------
__global__ void cvt_kernel(const float* __restrict__ in, bf16_t* __restrict__ out, size_t n) {
  for (size_t i = (size_t)blockIdx.x * 256 + threadIdx.x; i < n; i += (size_t)gridDim.x * 256)
    out[i] = (bf16_t)in[i];
}

__global__ void vtrans_kernel(const bf16_t* __restrict__ v, bf16_t* __restrict__ vt) {
  const size_t n = (size_t)Bc * Hc * DHc * Sc;
  for (size_t i = (size_t)blockIdx.x * 256 + threadIdx.x; i < n; i += (size_t)gridDim.x * 256) {
    const int s = (int)(i % Sc);
    size_t r = i / Sc;
    const int d = (int)(r % DHc); r /= DHc;
    const int h = (int)(r % Hc);
    const int b = (int)(r / Hc);
    vt[i] = v[((size_t)(b * Sc + s)) * Dc + h * DHc + d];
  }
}

// ---------------------------------------------------------------------------
// Orchestration
// ---------------------------------------------------------------------------
extern "C" void kernel_launch(void* const* d_in, const int* in_sizes, int n_in,
                              void* d_out, int out_size, void* d_ws, size_t ws_size,
                              hipStream_t stream) {
  (void)in_sizes; (void)n_in; (void)out_size; (void)ws_size;
  const float* x_in  = (const float*)d_in[0];
  const float* tau   = (const float*)d_in[1];
  const float* delta = (const float*)d_in[2];
  const float* Wq = (const float*)d_in[3];  const float* bq = (const float*)d_in[4];
  const float* Wk = (const float*)d_in[5];  const float* bk = (const float*)d_in[6];
  const float* Wv = (const float*)d_in[7];  const float* bv = (const float*)d_in[8];
  const float* Wo = (const float*)d_in[9];  const float* bo = (const float*)d_in[10];
  const float* W1 = (const float*)d_in[11]; const float* b1 = (const float*)d_in[12];
  const float* W2 = (const float*)d_in[13]; const float* b2 = (const float*)d_in[14];
  const float* g1 = (const float*)d_in[15]; const float* be1 = (const float*)d_in[16];
  const float* g2 = (const float*)d_in[17]; const float* be2 = (const float*)d_in[18];
  const float* gf = (const float*)d_in[19]; const float* bfi = (const float*)d_in[20];
  const float* Wp = (const float*)d_in[21]; const float* bp = (const float*)d_in[22];

  char* ws = (char*)d_ws;
  size_t off = 0;
  auto take = [&](size_t bytes) -> char* {
    char* p = ws + off;
    off += (bytes + 255) & ~(size_t)255;
    return p;
  };

  bf16_t* wq_bf  = (bf16_t*)take((size_t)Lc * Dc * Dc * 2);
  bf16_t* wk_bf  = (bf16_t*)take((size_t)Lc * Dc * Dc * 2);
  bf16_t* wv_bf  = (bf16_t*)take((size_t)Lc * Dc * Dc * 2);
  bf16_t* wo_bf  = (bf16_t*)take((size_t)Lc * Dc * Dc * 2);
  bf16_t* w1_bf  = (bf16_t*)take((size_t)Lc * Fc * Dc * 2);
  bf16_t* w2_bf  = (bf16_t*)take((size_t)Lc * Dc * Fc * 2);
  bf16_t* wp_bf  = (bf16_t*)take((size_t)Dc * Dc * 2);
  float*  x_f32  = (float*)take((size_t)Mrows * Dc * 4);
  bf16_t* x_bf   = (bf16_t*)take((size_t)Mrows * Dc * 2);
  bf16_t* q_bf   = (bf16_t*)take((size_t)Mrows * Dc * 2);
  bf16_t* k_bf   = (bf16_t*)take((size_t)Mrows * Dc * 2);
  bf16_t* v_bf   = (bf16_t*)take((size_t)Mrows * Dc * 2);
  bf16_t* vt_bf  = (bf16_t*)take((size_t)Mrows * Dc * 2);
  bf16_t* at_bf  = (bf16_t*)take((size_t)Mrows * Dc * 2);
  float*  y_f32  = (float*)take((size_t)Mrows * Dc * 4);
  bf16_t* h_bf   = (bf16_t*)take((size_t)Mrows * Fc * 2);

  const int CVG = 4096;
  cvt_kernel<<<CVG, 256, 0, stream>>>(Wq, wq_bf, (size_t)Lc * Dc * Dc);
  cvt_kernel<<<CVG, 256, 0, stream>>>(Wk, wk_bf, (size_t)Lc * Dc * Dc);
  cvt_kernel<<<CVG, 256, 0, stream>>>(Wv, wv_bf, (size_t)Lc * Dc * Dc);
  cvt_kernel<<<CVG, 256, 0, stream>>>(Wo, wo_bf, (size_t)Lc * Dc * Dc);
  cvt_kernel<<<CVG, 256, 0, stream>>>(W1, w1_bf, (size_t)Lc * Fc * Dc);
  cvt_kernel<<<CVG, 256, 0, stream>>>(W2, w2_bf, (size_t)Lc * Dc * Fc);
  cvt_kernel<<<CVG, 256, 0, stream>>>(Wp, wp_bf, (size_t)Dc * Dc);
  cvt_kernel<<<CVG, 256, 0, stream>>>(x_in, x_bf, (size_t)Mrows * Dc);

  const dim3 gDD(Dc / 128, Mrows / 128);  // 1024-wide GEMMs
  const dim3 gFD(Fc / 128, Mrows / 128);  // 4096-wide GEMMs

  for (int l = 0; l < Lc; l++) {
    const bf16_t* wq_l = wq_bf + (size_t)l * Dc * Dc;
    const bf16_t* wk_l = wk_bf + (size_t)l * Dc * Dc;
    const bf16_t* wv_l = wv_bf + (size_t)l * Dc * Dc;
    const bf16_t* wo_l = wo_bf + (size_t)l * Dc * Dc;
    const bf16_t* w1_l = w1_bf + (size_t)l * Fc * Dc;
    const bf16_t* w2_l = w2_bf + (size_t)l * Dc * Fc;

    gemm_kernel<1><<<gDD, 256, 0, stream>>>(x_bf, wq_l, bq + l * Dc, q_bf, Mrows, Dc, Dc);
    gemm_kernel<1><<<gDD, 256, 0, stream>>>(x_bf, wk_l, bk + l * Dc, k_bf, Mrows, Dc, Dc);
    gemm_kernel<1><<<gDD, 256, 0, stream>>>(x_bf, wv_l, bv + l * Dc, v_bf, Mrows, Dc, Dc);
    vtrans_kernel<<<CVG, 256, 0, stream>>>(v_bf, vt_bf);

    attn_kernel<<<Bc * Hc * (Sc / 32), 256, ATT_SMEM, stream>>>(q_bf, k_bf, vt_bf, tau,
                                                                delta, at_bf);

    gemm_kernel<0><<<gDD, 256, 0, stream>>>(at_bf, wo_l, bo + l * Dc, y_f32, Mrows, Dc, Dc);
    ln_kernel<<<Mrows, 256, 0, stream>>>(y_f32, (l == 0) ? x_in : x_f32,
                                         g1 + l * Dc, be1 + l * Dc, x_f32, x_bf);

    gemm_kernel<2><<<gFD, 256, 0, stream>>>(x_bf, w1_l, b1 + l * Fc, h_bf, Mrows, Fc, Dc);
    gemm_kernel<0><<<gDD, 256, 0, stream>>>(h_bf, w2_l, b2 + l * Dc, y_f32, Mrows, Dc, Fc);
    ln_kernel<<<Mrows, 256, 0, stream>>>(y_f32, x_f32, g2 + l * Dc, be2 + l * Dc, x_f32, x_bf);
  }

  // final LN (no residual) + output projection (fp32 -> d_out)
  ln_kernel<<<Mrows, 256, 0, stream>>>(x_f32, nullptr, gf, bfi, x_f32, x_bf);
  gemm_kernel<0><<<gDD, 256, 0, stream>>>(x_bf, wp_bf, bp, d_out, Mrows, Dc, Dc);
}
